// EquiMessagePsuedo2_34376918237209
// MI455X (gfx1250) — matrix operand order repaired and checked by reading, not compiled, not run on hardware
//
#include <hip/hip_runtime.h>
#include <math.h>

// ---------------------------------------------------------------------------
// EquiMessage (PaiNN-style) edge kernel for MI455X (gfx1250, wave32, WMMA).
// N=40000 nodes, E=400000 edges, F=64, N_RBF=20, CUTOFF=5.
//
// Per-edge einsums run as bf16 WMMA 16x16x32 tiles; edge weight matrices are
// staged into LDS by the Tensor Data Mover (overlapped with the cat-matrix
// build), fenced with s_wait_tensorcnt; segment-sum via global f32 atomics
// directly from WMMA D-fragments. Node data (~185MB) is L2-resident (192MB).
// ---------------------------------------------------------------------------

typedef __attribute__((ext_vector_type(16))) __bf16          v16bf;
typedef __attribute__((ext_vector_type(8)))  float           v8f;
typedef __attribute__((ext_vector_type(8)))  unsigned short  v8us;
typedef __attribute__((ext_vector_type(4)))  unsigned int    u32x4;
typedef __attribute__((ext_vector_type(4)))  int             i32x4;
typedef __attribute__((ext_vector_type(8)))  int             i32x8;

union FragBF { v16bf v; v8us p[2]; };

#if defined(__has_builtin)
#if __has_builtin(__builtin_amdgcn_tensor_load_to_lds) && __has_builtin(__builtin_amdgcn_s_wait_tensorcnt)
#define HAVE_TDM 1
#endif
#endif
#ifndef HAVE_TDM
#define HAVE_TDM 0
#endif

static __device__ __forceinline__ unsigned short f2bf(float x) {
    unsigned int u = __float_as_uint(x);
    u += 0x7FFFu + ((u >> 16) & 1u);          // round to nearest even
    return (unsigned short)(u >> 16);
}
static __device__ __forceinline__ float bf2f(unsigned short b) {
    return __uint_as_float(((unsigned int)b) << 16);
}
static __device__ __forceinline__ v8f vzero8() {
    v8f z;
#pragma unroll
    for (int i = 0; i < 8; i++) z[i] = 0.0f;
    return z;
}

#if HAVE_TDM
// Issue one TDM 2D tile load: rows x Kelems bf16, contiguous, global -> LDS.
// D# layout per CDNA5 ISA ch.8 (group0: count/lds_addr/global_addr/type,
// group1: data_size=2B, tensor_dim0=K, tensor_dim1=rows, tile_dim0=K,
// tile_dim1=rows, stride0=K; groups 2/3 zero for a 2D tensor).
static __device__ __forceinline__ void tdm_load_2d(unsigned lds_off,
                                                   const unsigned short* gsrc,
                                                   unsigned Kelems, unsigned rows) {
    unsigned long long ga = (unsigned long long)(uintptr_t)gsrc;
    u32x4 g0;
    g0[0] = 1u;                                            // count=1 (user D#)
    g0[1] = lds_off;                                       // lds_addr (bytes)
    g0[2] = (unsigned)(ga & 0xFFFFFFFFu);                  // global_addr[31:0]
    g0[3] = (unsigned)((ga >> 32) & 0x01FFFFFFu) | (2u << 30); // addr[56:32] | type=2
    i32x8 g1;
    g1[0] = (int)(1u << 16);                               // data_size=1 (2 bytes)
    g1[1] = (int)((Kelems & 0xFFFFu) << 16);               // tensor_dim0[15:0] @bit48
    g1[2] = (int)(((Kelems >> 16) & 0xFFFFu) | ((rows & 0xFFFFu) << 16)); // dim0 hi | dim1 lo
    g1[3] = (int)(((rows >> 16) & 0xFFFFu) | ((Kelems & 0xFFFFu) << 16)); // dim1 hi | tile_dim0
    g1[4] = (int)(rows & 0xFFFFu);                         // tile_dim1 (tile_dim2=0)
    g1[5] = (int)Kelems;                                   // tensor_dim0_stride[31:0]
    g1[6] = 0;
    g1[7] = 0;
    i32x4 z4; z4[0] = z4[1] = z4[2] = z4[3] = 0;
#if __clang_major__ >= 23
    i32x8 z8;
#pragma unroll
    for (int i = 0; i < 8; i++) z8[i] = 0;
    __builtin_amdgcn_tensor_load_to_lds(g0, g1, z4, z4, z8, 0);
#else
    __builtin_amdgcn_tensor_load_to_lds(g0, g1, z4, z4, 0);
#endif
}
#endif

// One 16x16 output tile: D = A(16xK, LDS bf16 row-major) * B(Kx16) + acc.
// B is read from [n][k] row-major storage (contiguous 16B per lane).
// A-frag (ISA 7.12.2, 16-bit A 16x32): lane l holds row M=l&15;
//   lane<16: elems 0..7 -> K=kt+0..7,  elems 8..15 -> K=kt+16..23
//   lane>=16: elems 0..7 -> K=kt+8..15, elems 8..15 -> K=kt+24..31
// B-frag: lane l holds col N=l&15; same K mapping as A-frag.
template <int K>
static __device__ __forceinline__ v8f run_tile(const unsigned short* cat,   // row stride K
                                               const unsigned short* wmat,  // [n][K] bf16
                                               int c, int n0, int ln, v8f acc) {
    const int mrow  = (c << 4) + (ln & 15);
    const int koffA = (ln >= 16) ? 8 : 0;
    const int koffB = (ln >= 16) ? 8 : 0;
    const unsigned short* arow = cat + (size_t)mrow * K;
    const unsigned short* brow = wmat + (size_t)(n0 + (ln & 15)) * K;
#pragma unroll
    for (int kt = 0; kt < K; kt += 32) {
        FragBF a, b;
        a.p[0] = *(const v8us*)(arow + kt + koffA);
        a.p[1] = *(const v8us*)(arow + kt + koffA + 16);
        b.p[0] = *(const v8us*)(brow + kt + koffB);
        b.p[1] = *(const v8us*)(brow + kt + koffB + 16);
        acc = __builtin_amdgcn_wmma_f32_16x16x32_bf16(
            false, a.v, false, b.v, (short)0, acc, false, false);
    }
    return acc;
}

// ---------------------------------------------------------------------------
// Kernel 0: convert weights f32 -> bf16 in workspace.
// ---------------------------------------------------------------------------
__global__ void prep_weights(const float* __restrict__ W1, const float* __restrict__ W2,
                             const float* __restrict__ Wdvbar, const float* __restrict__ Wdv,
                             unsigned short* __restrict__ w1t, unsigned short* __restrict__ w2t,
                             unsigned short* __restrict__ wdvb, unsigned short* __restrict__ wdv) {
    int t = blockIdx.x * blockDim.x + threadIdx.x;
    if (t < 4096) {                       // W1^T  [m][f]
        int m = t >> 6, f = t & 63;
        w1t[m * 64 + f] = f2bf(W1[f * 64 + m]);
    } else if (t < 20480) {               // W2^T  [m][f]
        int u = t - 4096; int m = u >> 6, f = u & 63;
        w2t[m * 64 + f] = f2bf(W2[f * 256 + m]);
    } else if (t < 36864) {               // W_dvbar [g][f] direct
        int u = t - 20480;
        wdvb[u] = f2bf(Wdvbar[u]);
    } else if (t < 49152) {               // W_dv [g][f] direct
        int u = t - 36864;
        wdv[u] = f2bf(Wdv[u]);
    }
}

__global__ void zero_out(float4* __restrict__ o, int n4) {
    for (int i = blockIdx.x * blockDim.x + threadIdx.x; i < n4; i += gridDim.x * blockDim.x)
        o[i] = make_float4(0.f, 0.f, 0.f, 0.f);
}

// ---------------------------------------------------------------------------
// Kernel 2: node MLP, h = silu(s@W1+b1)@W2+b2, WMMA tiled, 16 nodes/block.
// ---------------------------------------------------------------------------
__global__ __launch_bounds__(256) void node_mlp(const float* __restrict__ s,
                                                const float* __restrict__ b1,
                                                const float* __restrict__ b2,
                                                const unsigned short* __restrict__ w1t,
                                                const unsigned short* __restrict__ w2t,
                                                unsigned short* __restrict__ h, int N) {
    __shared__ __align__(16) unsigned short sS[16 * 64];
    __shared__ __align__(16) unsigned short sT[16 * 64];
    const int tid = threadIdx.x;
    const int n0  = blockIdx.x * 16;
    for (int idx = tid; idx < 1024; idx += 256) {
        int e = idx >> 6, f = idx & 63;
        int n = n0 + e; if (n >= N) n = N - 1;
        sS[idx] = f2bf(s[(size_t)n * 64 + f]);
    }
    __syncthreads();
    const int wv = tid >> 5, ln = tid & 31;
    if (wv < 4) {                                   // stage 1: t = silu(s@W1+b1)
        int nt = wv * 16;
        v8f acc = run_tile<64>(sS, w1t, 0, nt, ln, vzero8());
#pragma unroll
        for (int r = 0; r < 8; r++) {
            int e = r + ((ln >> 4) << 3);
            int m = nt + (ln & 15);
            float x = acc[r] + b1[m];
            float sig = __builtin_amdgcn_rcpf(1.f + __expf(-x));   // fast sigmoid
            sT[e * 64 + m] = f2bf(x * sig);
        }
    }
    __syncthreads();
    for (int jb = wv; jb < 16; jb += 8) {           // stage 2: h = t@W2+b2
        int m0 = jb * 16;
        v8f acc = run_tile<64>(sT, w2t, 0, m0, ln, vzero8());
#pragma unroll
        for (int r = 0; r < 8; r++) {
            int e = r + ((ln >> 4) << 3);
            int m = m0 + (ln & 15);
            if (n0 + e < N) h[(size_t)(n0 + e) * 256 + m] = f2bf(acc[r] + b2[m]);
        }
    }
}

// ---------------------------------------------------------------------------
// Kernel 3: fused edge kernel. 16 edges/block, 256 threads (8 wave32s).
// ---------------------------------------------------------------------------
__global__ __launch_bounds__(256) void edge_kernel(
    const float* __restrict__ s, const float* __restrict__ sbar,
    const float* __restrict__ v, const float* __restrict__ vbar,
    const float* __restrict__ r_ij, const int* __restrict__ nbrs,
    const float* __restrict__ Wd, const float* __restrict__ bd,
    const unsigned short* __restrict__ hws,
    const unsigned short* __restrict__ wdvb,   // [g][f] 64x256 bf16 (global)
    const unsigned short* __restrict__ wdv,    // [g][f] 64x192 bf16 (global)
    float* __restrict__ dh, float* __restrict__ dhbar,
    float* __restrict__ dv, float* __restrict__ dvbar, int E) {
    __shared__ __align__(16) unsigned short sWvb[64 * 256];      // staged W_dvbar
    __shared__ __align__(16) unsigned short sWv[64 * 192];       // staged W_dv
    __shared__ __align__(16) unsigned short sCatVbar[48 * 256];  // row = c*16+e
    __shared__ __align__(16) unsigned short sCatV[48 * 192];
    __shared__ float sInv[4 * 16 * 64];                          // [split][e][f]
    __shared__ float sRbf[16 * 20];
    __shared__ float sUnit[16 * 3];
    __shared__ float sEnv[16];
    __shared__ float sDist[16];
    __shared__ int   sI[16], sJ[16], sValid[16];

    const int tid = threadIdx.x;
    const int wv = tid >> 5, ln = tid & 31;
    const int e0  = blockIdx.x * 16;
    const float PI = 3.14159265358979323846f;

    // ---- Stage weights into LDS -------------------------------------------
#if HAVE_TDM
    if (wv == 0) {   // one wave issues the DMAs; they overlap P0..P2 below
        tdm_load_2d((unsigned)(uintptr_t)(void*)sWvb, wdvb, 256u, 64u);
        tdm_load_2d((unsigned)(uintptr_t)(void*)sWv,  wdv,  192u, 64u);
    }
#else
    {   // cooperative fallback copy (16B chunks)
        const uint4* g1 = (const uint4*)wdvb;
        uint4*       l1 = (uint4*)sWvb;
        for (int i = tid; i < 2048; i += 256) l1[i] = g1[i];
        const uint4* g2 = (const uint4*)wdv;
        uint4*       l2 = (uint4*)sWv;
        for (int i = tid; i < 1536; i += 256) l2[i] = g2[i];
    }
#endif

    // ---- P0a: per-edge geometry -------------------------------------------
    if (tid < 16) {
        int eg  = e0 + tid;
        int val = (eg < E) ? 1 : 0;
        int egc = val ? eg : 0;
        int ni = nbrs[2 * egc], nj = nbrs[2 * egc + 1];
        float rx = r_ij[3 * egc + 0], ry = r_ij[3 * egc + 1], rz = r_ij[3 * egc + 2];
        float d  = sqrtf(rx * rx + ry * ry + rz * rz + 3e-8f);
        float id = 1.f / d;
        sI[tid] = ni; sJ[tid] = nj; sValid[tid] = val; sDist[tid] = d;
        sUnit[tid * 3 + 0] = rx * id;
        sUnit[tid * 3 + 1] = ry * id;
        sUnit[tid * 3 + 2] = rz * id;
        float env = (d < 5.0f) ? 0.5f * (__cosf(PI * d * 0.2f) + 1.0f) : 0.0f;
        sEnv[tid] = val ? env : 0.0f;
    }
    __syncthreads();
    // ---- P0b: RBF table ----------------------------------------------------
    for (int idx = tid; idx < 320; idx += 256) {
        int e = idx / 20, k = idx % 20;
        float d = sDist[e];
        sRbf[idx] = __sinf((float)(k + 1) * PI * d * 0.2f) / d;
    }
    __syncthreads();
    // ---- P1: w_s = (rbf@Wd+bd)*env ; inv = phi*w_s ------------------------
    {
        int m = tid, sp = m >> 6, f = m & 63;
        for (int e = 0; e < 16; e++) {
            float acc = bd[m];
#pragma unroll
            for (int k = 0; k < 20; k++) acc += sRbf[e * 20 + k] * Wd[k * 256 + m];
            float w   = acc * sEnv[e];
            float phi = bf2f(hws[(size_t)sJ[e] * 256 + m]);
            sInv[(sp * 16 + e) * 64 + f] = phi * w;
        }
    }
    __syncthreads();
    // ---- P2: build bf16 cat matrices + dh/dhbar atomics -------------------
    for (int it = 0; it < 4; it++) {
        int idx = it * 256 + tid;
        int e = idx >> 6, f = idx & 63;
        int ni = sI[e], nj = sJ[e];
        const float* pvi  = v    + ((size_t)ni * 64 + f) * 3;
        const float* pvj  = v    + ((size_t)nj * 64 + f) * 3;
        const float* pvbi = vbar + ((size_t)ni * 64 + f) * 3;
        const float* pvbj = vbar + ((size_t)nj * 64 + f) * 3;
        float vi_[3], vj_[3], vbi_[3], vbj_[3];
#pragma unroll
        for (int c = 0; c < 3; c++) {
            vi_[c] = pvi[c]; vj_[c] = pvj[c]; vbi_[c] = pvbi[c]; vbj_[c] = pvbj[c];
        }
        float sif  = s[(size_t)ni * 64 + f];
        float sbif = sbar[(size_t)ni * 64 + f];
        float cij[3], cbb[3], civb[3];
        cij[0]  = vi_[1] * vj_[2]  - vi_[2] * vj_[1];
        cij[1]  = vi_[2] * vj_[0]  - vi_[0] * vj_[2];
        cij[2]  = vi_[0] * vj_[1]  - vi_[1] * vj_[0];
        cbb[0]  = vbi_[1] * vbj_[2] - vbi_[2] * vbj_[1];
        cbb[1]  = vbi_[2] * vbj_[0] - vbi_[0] * vbj_[2];
        cbb[2]  = vbi_[0] * vbj_[1] - vbi_[1] * vbj_[0];
        civb[0] = vi_[1] * vbj_[2] - vi_[2] * vbj_[1];
        civb[1] = vi_[2] * vbj_[0] - vi_[0] * vbj_[2];
        civb[2] = vi_[0] * vbj_[1] - vi_[1] * vbj_[0];
#pragma unroll
        for (int c = 0; c < 3; c++) {
            int row = c * 16 + e;
            sCatVbar[row * 256 + f]       = f2bf(sif * vbj_[c]);   // si * vbar_j
            sCatVbar[row * 256 + 64 + f]  = f2bf(sbif * vj_[c]);   // sbar_i * v_j
            sCatVbar[row * 256 + 128 + f] = f2bf(cij[c]);          // vi x vj
            sCatVbar[row * 256 + 192 + f] = f2bf(cbb[c]);          // vbi x vbj
            sCatV[row * 192 + f]          = f2bf(sif * vj_[c]);    // si * v_j
            sCatV[row * 192 + 64 + f]     = f2bf(sbif * vbj_[c]);  // sbar_i * vbar_j
            sCatV[row * 192 + 128 + f]    = f2bf(civb[c]);         // vi x vbj
        }
        float dot = vi_[0] * vbj_[0] + vi_[1] * vbj_[1] + vi_[2] * vbj_[2];
        if (sValid[e]) {
            atomicAdd(&dh[(size_t)ni * 64 + f],    sInv[(1 * 16 + e) * 64 + f]);
            atomicAdd(&dhbar[(size_t)ni * 64 + f], sInv[(3 * 16 + e) * 64 + f] * dot);
        }
    }
#if HAVE_TDM
    if (wv == 0) __builtin_amdgcn_s_wait_tensorcnt(0);   // weights landed in LDS
#endif
    __syncthreads();
    // ---- P3: WMMA tile jobs (12 dvbar + 12 dv) + scatter-add --------------
    for (int jb = wv; jb < 24; jb += 8) {
        const bool isV = (jb >= 12);
        const int  t  = isV ? (jb - 12) : jb;
        const int  c  = t >> 2;
        const int  n0 = (t & 3) << 4;
        if (!isV) {
            v8f acc = run_tile<256>(sCatVbar, sWvb, c, n0, ln, vzero8());
#pragma unroll
            for (int r = 0; r < 8; r++) {
                int e = r + ((ln >> 4) << 3);
                int g = n0 + (ln & 15);
                if (sValid[e])
                    atomicAdd(&dvbar[((size_t)sI[e] * 64 + g) * 3 + c], acc[r]);
            }
        } else {
            // fold elementwise part (split2*unit + split0*v_j) into the C operand
            v8f acc;
#pragma unroll
            for (int r = 0; r < 8; r++) {
                int e = r + ((ln >> 4) << 3);
                int g = n0 + (ln & 15);
                acc[r] = sInv[(2 * 16 + e) * 64 + g] * sUnit[e * 3 + c]
                       + sInv[(0 * 16 + e) * 64 + g] * v[((size_t)sJ[e] * 64 + g) * 3 + c];
            }
            acc = run_tile<192>(sCatV, sWv, c, n0, ln, acc);
#pragma unroll
            for (int r = 0; r < 8; r++) {
                int e = r + ((ln >> 4) << 3);
                int g = n0 + (ln & 15);
                if (sValid[e])
                    atomicAdd(&dv[((size_t)sI[e] * 64 + g) * 3 + c], acc[r]);
            }
        }
    }
}

// ---------------------------------------------------------------------------
extern "C" void kernel_launch(void* const* d_in, const int* in_sizes, int n_in,
                              void* d_out, int out_size, void* d_ws, size_t ws_size,
                              hipStream_t stream) {
    const float* s      = (const float*)d_in[0];
    const float* sbar   = (const float*)d_in[1];
    const float* v      = (const float*)d_in[2];
    const float* vbar   = (const float*)d_in[3];
    const float* rij    = (const float*)d_in[4];
    const int*   nbrs   = (const int*)d_in[5];
    const float* W1     = (const float*)d_in[6];
    const float* b1     = (const float*)d_in[7];
    const float* W2     = (const float*)d_in[8];
    const float* b2     = (const float*)d_in[9];
    const float* Wd     = (const float*)d_in[10];
    const float* bd     = (const float*)d_in[11];
    const float* Wdvbar = (const float*)d_in[12];
    const float* Wdv    = (const float*)d_in[13];

    const int N = in_sizes[0] / 64;   // 40000
    const int E = in_sizes[4] / 3;    // 400000

    float* out   = (float*)d_out;
    float* dh    = out;
    float* dhbar = out + (size_t)N * 64;
    float* dvp   = out + (size_t)N * 128;
    float* dvbp  = out + (size_t)N * 128 + (size_t)N * 192;

    // workspace layout (bf16/ushort): h (N*256), W1^T, W2^T, W_dvbar, W_dv
    unsigned short* hws  = (unsigned short*)d_ws;
    unsigned short* w1t  = hws + (size_t)N * 256;
    unsigned short* w2t  = w1t + 64 * 64;
    unsigned short* wdvb = w2t + 256 * 64;
    unsigned short* wdvw = wdvb + 64 * 256;

    prep_weights<<<(49152 + 255) / 256, 256, 0, stream>>>(W1, W2, Wdvbar, Wdv,
                                                          w1t, w2t, wdvb, wdvw);
    int n4 = out_size / 4;
    zero_out<<<4096, 256, 0, stream>>>((float4*)d_out, n4);
    node_mlp<<<(N + 15) / 16, 256, 0, stream>>>(s, b1, b2, w1t, w2t, hws, N);
    edge_kernel<<<(E + 15) / 16, 256, 0, stream>>>(s, sbar, v, vbar, rij, nbrs,
                                                   Wd, bd, hws, wdvb, wdvw,
                                                   dh, dhbar, dvp, dvbp, E);
}
